// SelfAttention_19997367730626
// MI455X (gfx1250) — matrix operand order repaired and checked by reading, not compiled, not run on hardware
//
#include <hip/hip_runtime.h>
#include <math.h>

typedef __attribute__((ext_vector_type(16))) __bf16 v16bf;
typedef __attribute__((ext_vector_type(8)))  __bf16 bf16x8;
typedef __attribute__((ext_vector_type(4)))  __bf16 bf16x4;
typedef __attribute__((ext_vector_type(8)))  float  v8f;

// Problem geometry is fixed by the reference -> compile-time constants so the
// backend folds row strides into immediate offsets (kills the u64 add chains).
#define DD 1024   // model dim
#define SS 2048   // sequence length
#define TM 128    // block M tile
#define TN 64     // block N tile (qkv / attnv)
#define TK 32     // WMMA K step (bf16)
#define TKK 64    // K chunk per LDS stage (2 WMMA steps per barrier)
#define LDT 72    // LDS row stride in bf16 (64 + 8 pad, keeps 16B alignment)

__device__ __forceinline__ __bf16 f2bf(float f) { return (__bf16)f; }

// A fragment (16x32 bf16): lane = M row (lane&15); h=lane>>4 selects K groups
// [h*8, h*8+8) in VGPRs 0-3 and [16+h*8, 16+h*8+8) in VGPRs 4-7.
__device__ __forceinline__ v16bf load_fragA(const __bf16* base, int ld) {
    const int lane = threadIdx.x & 31;
    const __bf16* p = base + (size_t)(lane & 15) * ld + ((lane >> 4) << 3);
    v16bf out;
    ((bf16x8*)&out)[0] = *(const bf16x8*)(p);
    ((bf16x8*)&out)[1] = *(const bf16x8*)(p + 16);
    return out;
}

// B fragment (32x16 bf16) from a [N][K] (K-contiguous) tile: lane = N column
// (lane&15); h=lane>>4 selects K [h*8, h*8+8) then [16+h*8, 16+h*8+8).
__device__ __forceinline__ v16bf load_fragB(const __bf16* base, int ld) {
    const int lane = threadIdx.x & 31;
    const __bf16* p = base + (size_t)(lane & 15) * ld + ((lane >> 4) << 3);
    v16bf out;
    ((bf16x8*)&out)[0] = *(const bf16x8*)(p);
    ((bf16x8*)&out)[1] = *(const bf16x8*)(p + 16);
    return out;
}

__device__ __forceinline__ v8f wmma_bf16(v16bf a, v16bf b, v8f c) {
    return __builtin_amdgcn_wmma_f32_16x16x32_bf16(false, a, false, b,
                                                   (short)0, c, false, false);
}

// ---------------------------------------------------------------------------
// Kernel 1: Y[r, e] = sum_d x[b, d, s] * W[e, d] + bias[e],  r = b*SS + s
// K=64 chunks, register-staged double buffering: next chunk's global loads
// issue while current chunk's WMMAs run.
// ---------------------------------------------------------------------------
__device__ __forceinline__ void qkv_load(const float* __restrict__ xb,
                                         const float* __restrict__ W,
                                         int k0, int s_base, int n_blk,
                                         int da, int sg, int eb, int kg,
                                         float4 ra[8], float4 rb[4]) {
    const float* xp = xb + (size_t)(k0 + da) * SS + s_base + sg;
#pragma unroll
    for (int i = 0; i < 8; ++i) ra[i] = *(const float4*)(xp + i * 16);
    const float* wp = W + (size_t)(n_blk + eb) * DD + k0 + kg;
#pragma unroll
    for (int i = 0; i < 4; ++i) rb[i] = *(const float4*)(wp + i * 16);
}

__global__ __launch_bounds__(256) void qkv_gemm(
    const float* __restrict__ x, const float* __restrict__ W,
    const float* __restrict__ bias, __bf16* __restrict__ Y) {
    __shared__ __align__(16) __bf16 lds_a[TM * LDT];  // [m][k]
    __shared__ __align__(16) __bf16 lds_b[TN * LDT];  // [n][k]
    const int tid = threadIdx.x, lane = tid & 31, wid = tid >> 5;
    const int m_blk = blockIdx.y * TM;   // flat row over B*SS (SS % TM == 0)
    const int n_blk = blockIdx.x * TN;
    const int b = m_blk / SS, s_base = m_blk % SS;
    const float* xb = x + (size_t)b * DD * SS;
    const int wm = (wid & 3) * 32, wn = (wid >> 2) * 32;

    // A stager: 64 d-rows x 4 threads, 8 float4 each (covers 128 s)
    const int da = tid >> 2, sg = (tid & 3) * 4;
    // B stager: 64 e-rows x 4 threads, 4 float4 each (covers 64 k)
    const int eb = tid >> 2, kg = (tid & 3) * 4;

    v8f acc[2][2] = {};
    float4 ra[8], rb[4];
    qkv_load(xb, W, 0, s_base, n_blk, da, sg, eb, kg, ra, rb);

    for (int k0 = 0; k0 < DD; k0 += TKK) {
        // ---- regs -> LDS (A transposed: x[d][s] -> lds_a[s][d]) ----
#pragma unroll
        for (int i = 0; i < 8; ++i) {
            int sl = sg + i * 16;
            lds_a[(sl + 0) * LDT + da] = f2bf(ra[i].x);
            lds_a[(sl + 1) * LDT + da] = f2bf(ra[i].y);
            lds_a[(sl + 2) * LDT + da] = f2bf(ra[i].z);
            lds_a[(sl + 3) * LDT + da] = f2bf(ra[i].w);
        }
#pragma unroll
        for (int i = 0; i < 4; ++i) {
            bf16x4 t;
            t[0] = f2bf(rb[i].x); t[1] = f2bf(rb[i].y);
            t[2] = f2bf(rb[i].z); t[3] = f2bf(rb[i].w);
            *(bf16x4*)&lds_b[eb * LDT + kg + i * 16] = t;
        }
        __syncthreads();
        if (k0 + TKK < DD)   // prefetch next chunk into regs while WMMAs run
            qkv_load(xb, W, k0 + TKK, s_base, n_blk, da, sg, eb, kg, ra, rb);
#pragma unroll
        for (int kk = 0; kk < TKK; kk += TK) {
            v16bf a0 = load_fragA(&lds_a[(wm +  0) * LDT + kk], LDT);
            v16bf a1 = load_fragA(&lds_a[(wm + 16) * LDT + kk], LDT);
            v16bf b0 = load_fragB(&lds_b[(wn +  0) * LDT + kk], LDT);
            v16bf b1 = load_fragB(&lds_b[(wn + 16) * LDT + kk], LDT);
            acc[0][0] = wmma_bf16(a0, b0, acc[0][0]);
            acc[0][1] = wmma_bf16(a0, b1, acc[0][1]);
            acc[1][0] = wmma_bf16(a1, b0, acc[1][0]);
            acc[1][1] = wmma_bf16(a1, b1, acc[1][1]);
        }
        __syncthreads();
    }
    const int nl = lane & 15, mh = (lane >> 4) << 3;
#pragma unroll
    for (int mi = 0; mi < 2; ++mi)
#pragma unroll
        for (int ni = 0; ni < 2; ++ni) {
            int ncol = n_blk + wn + ni * 16 + nl;
            float bv = bias[ncol];
            __bf16* yp = Y + (size_t)(m_blk + wm + mi * 16 + mh) * DD + ncol;
#pragma unroll
            for (int v = 0; v < 8; ++v)
                yp[v * DD] = f2bf(acc[mi][ni][v] + bv);
        }
}

// ---------------------------------------------------------------------------
// Kernel 2: P[b, s, t] = (Q[b,s,:] . K[b,t,:]) * scale
// Direct global b128 fragment loads (Q/K are L2-resident); 32x64 wave tile
// (8 WMMA per 6 fragment loads) + lane-spread prefetch of next K tile.
// ---------------------------------------------------------------------------
__global__ __launch_bounds__(256) void score_gemm(
    const __bf16* __restrict__ Q, const __bf16* __restrict__ Km,
    __bf16* __restrict__ P, float scale) {
    const int tid = threadIdx.x, lane = tid & 31, wid = tid >> 5;
    const int m_blk = blockIdx.y * TM, n_blk = blockIdx.x * 128, bz = blockIdx.z;
    const __bf16* Qb = Q + (size_t)bz * SS * DD;
    const __bf16* Kb = Km + (size_t)bz * SS * DD;
    __bf16* Pb = P + (size_t)bz * SS * SS;
    const int wm = (wid & 3) * 32, wn = (wid >> 2) * 64;
    v8f acc[2][4] = {};
    for (int k0 = 0; k0 < DD; k0 += TK) {
        const __bf16* qa = Qb + (size_t)(m_blk + wm) * DD + k0;
        v16bf a0 = load_fragA(qa, DD);
        v16bf a1 = load_fragA(qa + 16 * DD, DD);
        v16bf bb[4];
#pragma unroll
        for (int ni = 0; ni < 4; ++ni)
            bb[ni] = load_fragB(Kb + (size_t)(n_blk + wn + ni * 16) * DD + k0, DD);
        if (k0 + TK < DD) {   // one 64B line per row = next K=32 slab
            __builtin_prefetch(Qb + (size_t)(m_blk + wm + lane) * DD + k0 + TK, 0, 3);
            __builtin_prefetch(Kb + (size_t)(n_blk + wn + lane) * DD + k0 + TK, 0, 3);
            __builtin_prefetch(Kb + (size_t)(n_blk + wn + 32 + lane) * DD + k0 + TK, 0, 3);
        }
#pragma unroll
        for (int ni = 0; ni < 4; ++ni) {
            acc[0][ni] = wmma_bf16(a0, bb[ni], acc[0][ni]);
            acc[1][ni] = wmma_bf16(a1, bb[ni], acc[1][ni]);
        }
    }
    const int nl = lane & 15, mh = (lane >> 4) << 3;
#pragma unroll
    for (int mi = 0; mi < 2; ++mi)
#pragma unroll
        for (int ni = 0; ni < 4; ++ni) {
            int col = n_blk + wn + ni * 16 + nl;
            int row = m_blk + wm + mi * 16 + mh;
            __bf16* pp = Pb + (size_t)row * SS + col;
#pragma unroll
            for (int v = 0; v < 8; ++v)
                pp[v * SS] = f2bf(acc[mi][ni][v] * scale);
        }
}

// ---------------------------------------------------------------------------
// Kernel 3: in-place row softmax over P (one block per row, f32 math).
// ---------------------------------------------------------------------------
__global__ __launch_bounds__(256) void softmax_rows(__bf16* __restrict__ P) {
    __shared__ float red[256];
    const int tid = threadIdx.x;
    __bf16* row = P + (size_t)blockIdx.x * SS;   // 2048 -> 8 per thread
    bf16x8 d = *(const bf16x8*)(row + tid * 8);
    float f[8];
    float lmax = -3.4e38f;
#pragma unroll
    for (int i = 0; i < 8; ++i) { f[i] = (float)d[i]; lmax = fmaxf(lmax, f[i]); }
    red[tid] = lmax; __syncthreads();
    for (int off = 128; off > 0; off >>= 1) {
        if (tid < off) red[tid] = fmaxf(red[tid], red[tid + off]);
        __syncthreads();
    }
    float rmax = red[0]; __syncthreads();
    float lsum = 0.f;
#pragma unroll
    for (int i = 0; i < 8; ++i) { f[i] = __expf(f[i] - rmax); lsum += f[i]; }
    red[tid] = lsum; __syncthreads();
    for (int off = 128; off > 0; off >>= 1) {
        if (tid < off) red[tid] += red[tid + off];
        __syncthreads();
    }
    float inv = 1.0f / red[0];
    bf16x8 o;
#pragma unroll
    for (int i = 0; i < 8; ++i) o[i] = f2bf(f[i] * inv);
    *(bf16x8*)(row + tid * 8) = o;
}

// ---------------------------------------------------------------------------
// Kernel 4: out[b, s, e] = sum_t P[b,s,t] * V[b,t,e]  (f32 output).
// P feeds A fragments straight from global; V chunk (64t x 64e) is
// register-staged then transposed into LDS as [e][t].
// ---------------------------------------------------------------------------
__global__ __launch_bounds__(256) void attnv_gemm(
    const __bf16* __restrict__ P, const __bf16* __restrict__ V,
    float* __restrict__ out) {
    __shared__ __align__(16) __bf16 lds_v[TN * LDT];  // [e][t]
    const int tid = threadIdx.x, lane = tid & 31, wid = tid >> 5;
    const int m_blk = blockIdx.y * TM, n_blk = blockIdx.x * TN, bz = blockIdx.z;
    const __bf16* Pb = P + (size_t)bz * SS * SS;
    const __bf16* Vb = V + (size_t)bz * SS * DD;
    float* ob = out + (size_t)bz * SS * DD;
    const int wm = (wid & 3) * 32, wn = (wid >> 2) * 32;
    // V stager: 64 t-rows x 4 threads, 2 x bf16x8 each (covers 64 e)
    const int tl = tid >> 2, eg = (tid & 3) * 8;
    v8f acc[2][2] = {};
    bf16x8 rv[2];
#pragma unroll
    for (int i = 0; i < 2; ++i)
        rv[i] = *(const bf16x8*)(Vb + (size_t)tl * DD + n_blk + eg + i * 32);

    for (int k0 = 0; k0 < SS; k0 += TKK) {
        // regs -> LDS transposed: V[t][e] -> lds_v[e][t]
#pragma unroll
        for (int i = 0; i < 2; ++i)
#pragma unroll
            for (int j = 0; j < 8; ++j)
                lds_v[(eg + i * 32 + j) * LDT + tl] = rv[i][j];
        __syncthreads();
        if (k0 + TKK < SS) {
#pragma unroll
            for (int i = 0; i < 2; ++i)
                rv[i] = *(const bf16x8*)(Vb + (size_t)(k0 + TKK + tl) * DD +
                                         n_blk + eg + i * 32);
            __builtin_prefetch(Pb + (size_t)(m_blk + wm + lane) * SS + k0 + TKK, 0, 3);
        }
#pragma unroll
        for (int kk = 0; kk < TKK; kk += TK) {
            v16bf a0 = load_fragA(Pb + (size_t)(m_blk + wm +  0) * SS + k0 + kk, SS);
            v16bf a1 = load_fragA(Pb + (size_t)(m_blk + wm + 16) * SS + k0 + kk, SS);
            v16bf b0 = load_fragB(&lds_v[(wn +  0) * LDT + kk], LDT);
            v16bf b1 = load_fragB(&lds_v[(wn + 16) * LDT + kk], LDT);
            acc[0][0] = wmma_bf16(a0, b0, acc[0][0]);
            acc[0][1] = wmma_bf16(a0, b1, acc[0][1]);
            acc[1][0] = wmma_bf16(a1, b0, acc[1][0]);
            acc[1][1] = wmma_bf16(a1, b1, acc[1][1]);
        }
        __syncthreads();
    }
    const int nl = lane & 15, mh = (lane >> 4) << 3;
#pragma unroll
    for (int mi = 0; mi < 2; ++mi)
#pragma unroll
        for (int ni = 0; ni < 2; ++ni) {
            int col = n_blk + wn + ni * 16 + nl;
            int row = m_blk + wm + mi * 16 + mh;
            float* pp = ob + (size_t)row * DD + col;
#pragma unroll
            for (int v = 0; v < 8; ++v)
                pp[v * DD] = acc[mi][ni][v];
        }
}

// ---------------------------------------------------------------------------
extern "C" void kernel_launch(void* const* d_in, const int* in_sizes, int n_in,
                              void* d_out, int out_size, void* d_ws, size_t ws_size,
                              hipStream_t stream) {
    const float* x  = (const float*)d_in[0];
    const float* Wq = (const float*)d_in[1];
    const float* bq = (const float*)d_in[2];
    const float* Wk = (const float*)d_in[3];
    const float* bk = (const float*)d_in[4];
    const float* Wv = (const float*)d_in[5];
    const float* bv = (const float*)d_in[6];
    const int B = in_sizes[0] / (DD * SS);     // 4
    const size_t BSD = (size_t)B * SS * DD;

    // Workspace layout (bf16): Q | K | V | P  == 16+16+16+32 MB = 80 MB
    __bf16* Q = (__bf16*)d_ws;
    __bf16* K = Q + BSD;
    __bf16* V = K + BSD;
    __bf16* P = V + BSD;                       // B*SS*SS
    float* out = (float*)d_out;

    dim3 blk(256);
    dim3 g1(DD / TN, (B * SS) / TM);
    qkv_gemm<<<g1, blk, 0, stream>>>(x, Wq, bq, Q);
    qkv_gemm<<<g1, blk, 0, stream>>>(x, Wk, bk, K);
    qkv_gemm<<<g1, blk, 0, stream>>>(x, Wv, bv, V);

    float scale = 1.0f / sqrtf((float)DD);
    dim3 g2(SS / 128, SS / TM, B);
    score_gemm<<<g2, blk, 0, stream>>>(Q, K, P, scale);

    softmax_rows<<<dim3(B * SS), blk, 0, stream>>>(P);

    dim3 g4(DD / TN, SS / TM, B);
    attnv_gemm<<<g4, blk, 0, stream>>>(P, V, out);
}
